// CRF_51015621542595
// MI455X (gfx1250) — compile-verified, matrix operand order checked by reading, not run on hardware
//
#include <hip/hip_runtime.h>
#include <hip/hip_fp16.h>
#include <math.h>

typedef __attribute__((ext_vector_type(16))) _Float16 v16h_t;
typedef __attribute__((ext_vector_type(8)))  _Float16 v8h_t;
typedef __attribute__((ext_vector_type(8)))  float    v8f_t;

#define B_ 512
#define S_ 512
#define T_ 128

#define ET_STRIDE 130   // halves per row of exp-transitions in LDS (even)
#define AE_STRIDE 136   // halves per row of exp-alpha in LDS (272B = 16B multiple)
#define AL_STRIDE 132   // floats per row of alpha in LDS

// ---------------------------------------------------------------------------
// Forward algorithm: log-partition per batch row via exp-space WMMA GEMM.
// Grid: B/16 blocks x 256 threads (8 waves). Wave w owns state columns
// [16w, 16w+16). Block owns batch rows [16*blockIdx.x, +16).
// ---------------------------------------------------------------------------
__global__ __launch_bounds__(256) void crf_forward_kernel(
    const float* __restrict__ emissions,     // [B][S][T]
    const float* __restrict__ transitions,   // [T][T]
    const float* __restrict__ start_t,       // [T]
    const float* __restrict__ end_t,         // [T]
    const int*   __restrict__ lengths,       // [B]
    float*       __restrict__ log_den)       // [B]
{
    __shared__ __align__(16) _Float16 Eh[T_ * ET_STRIDE];   // exp(trans - cmax[col])
    __shared__ __align__(16) _Float16 Ax[16 * AE_STRIDE];   // exp(alpha - rowmax)
    __shared__ float alpha_s[16 * AL_STRIDE];               // running log-alpha
    __shared__ float cmax[T_];
    __shared__ float rowmax[16];
    __shared__ float pred[256];                             // partial reductions
    __shared__ int   lenS[16];

    const int tid   = threadIdx.x;
    const int lane  = tid & 31;
    const int wave  = tid >> 5;            // N-tile id, 0..7
    const int bbase = blockIdx.x * 16;

    // --- 1) column max of transitions -------------------------------------
    if (tid < T_) {
        float mx = -INFINITY;
        for (int i = 0; i < T_; ++i) mx = fmaxf(mx, transitions[i * T_ + tid]);
        cmax[tid] = mx;
    }
    if (tid < 16) lenS[tid] = lengths[bbase + tid];
    __syncthreads();

    // --- 2) Eh = exp(trans - cmax) in f16 (values in (0,1]) ---------------
    for (int idx = tid; idx < T_ * T_; idx += 256) {
        int i = idx >> 7, j = idx & (T_ - 1);
        Eh[i * ET_STRIDE + j] = (_Float16)__expf(transitions[idx] - cmax[j]);
    }
    __syncthreads();

    // --- 3) loop-invariant B fragments (held in VGPRs for all 511 steps) --
    // B layout (32x16 f16, D=A*B): lane holds column n=lane%16; K packing:
    // lanes 0-15: halves 0-7 = K kb..kb+7, halves 8-15 = K kb+16..kb+23;
    // lanes 16-31: same with kb += 8.
    const int jcol = wave * 16 + (lane & 15);
    const int koff = (lane >= 16) ? 8 : 0;
    v16h_t Bf[4];
#pragma unroll
    for (int kc = 0; kc < 4; ++kc) {
        const int kb = kc * 32 + koff;
        v16h_t b;
#pragma unroll
        for (int h = 0; h < 8; ++h) {
            b[h]     = Eh[(kb + h)      * ET_STRIDE + jcol];
            b[8 + h] = Eh[(kb + 16 + h) * ET_STRIDE + jcol];
        }
        Bf[kc] = b;
    }

    // --- 4) alpha at t=0 ---------------------------------------------------
    for (int idx = tid; idx < 16 * T_; idx += 256) {
        int r = idx >> 7, c = idx & (T_ - 1);
        alpha_s[r * AL_STRIDE + c] =
            start_t[c] + emissions[((size_t)(bbase + r) * S_) * T_ + c];
    }
    __syncthreads();

    // D-tile lane mapping (f32 16x16 C/D): col = lane%16, rows r + 8*(lane>=16)
    const int drow0 = (lane >= 16) ? 8 : 0;
    const int dcol  = lane & 15;
    const int arow  = lane & 15;
    const int prow  = tid >> 4;            // renorm mapping: 16 threads/row
    const int pseg  = tid & 15;

    for (int t = 1; t < S_; ++t) {
        // ---- renormalize current alpha: rowmax + Ax = exp(alpha - m) -----
        float pm = -INFINITY;
#pragma unroll
        for (int c8 = 0; c8 < 8; ++c8)
            pm = fmaxf(pm, alpha_s[prow * AL_STRIDE + pseg * 8 + c8]);
        pred[prow * 16 + pseg] = pm;
        __syncthreads();
        if (tid < 16) {
            float m = pred[tid * 16];
            for (int s = 1; s < 16; ++s) m = fmaxf(m, pred[tid * 16 + s]);
            rowmax[tid] = m;
        }
        __syncthreads();
#pragma unroll
        for (int c8 = 0; c8 < 8; ++c8) {
            const int c = pseg * 8 + c8;
            Ax[prow * AE_STRIDE + c] =
                (_Float16)__expf(alpha_s[prow * AL_STRIDE + c] - rowmax[prow]);
        }
        __syncthreads();

        if (t + 1 < S_)  // warm L2 for next step's emissions tile
            __builtin_prefetch(
                &emissions[((size_t)(bbase + drow0) * S_ + (t + 1)) * T_ + wave * 16], 0, 1);

        // ---- A fragments (two 16B LDS loads each) + 4 WMMAs over K=128 ---
        v8f_t acc = {};
#pragma unroll
        for (int kc = 0; kc < 4; ++kc) {
            const int kb = kc * 32 + koff;
            const v8h_t lo = *(const v8h_t*)&Ax[arow * AE_STRIDE + kb];
            const v8h_t hi = *(const v8h_t*)&Ax[arow * AE_STRIDE + kb + 16];
            v16h_t a;
#pragma unroll
            for (int h = 0; h < 8; ++h) { a[h] = lo[h]; a[8 + h] = hi[h]; }
            acc = __builtin_amdgcn_wmma_f32_16x16x32_f16(
                false, a, false, Bf[kc], (short)0, acc, false, false);
        }

        // ---- alpha' = m + cmax + emit + log(acc), masked by length -------
#pragma unroll
        for (int r = 0; r < 8; ++r) {
            const int row = drow0 + r;
            const int col = wave * 16 + dcol;
            const float oldv = alpha_s[row * AL_STRIDE + col];
            const float emit =
                emissions[((size_t)(bbase + row) * S_ + t) * T_ + col];
            const float nv =
                rowmax[row] + cmax[col] + emit + __logf((float)acc[r]);
            alpha_s[row * AL_STRIDE + col] = (t < lenS[row]) ? nv : oldv;
        }
        __syncthreads();
    }

    // --- final: log_den[b] = logsumexp_j(alpha[b,j] + end_t[j]) ------------
    float pm = -INFINITY;
#pragma unroll
    for (int c8 = 0; c8 < 8; ++c8) {
        const int c = pseg * 8 + c8;
        pm = fmaxf(pm, alpha_s[prow * AL_STRIDE + c] + end_t[c]);
    }
    pred[prow * 16 + pseg] = pm;
    __syncthreads();
    if (tid < 16) {
        float m = pred[tid * 16];
        for (int s = 1; s < 16; ++s) m = fmaxf(m, pred[tid * 16 + s]);
        rowmax[tid] = m;
    }
    __syncthreads();
    float ps = 0.f;
#pragma unroll
    for (int c8 = 0; c8 < 8; ++c8) {
        const int c = pseg * 8 + c8;
        ps += __expf(alpha_s[prow * AL_STRIDE + c] + end_t[c] - rowmax[prow]);
    }
    pred[prow * 16 + pseg] = ps;
    __syncthreads();
    if (tid < 16) {
        float s = 0.f;
        for (int k = 0; k < 16; ++k) s += pred[tid * 16 + k];
        log_den[bbase + tid] = rowmax[tid] + __logf(s);
    }
}

// ---------------------------------------------------------------------------
// Numerator: gold-path score, one thread per batch row.
// ---------------------------------------------------------------------------
__global__ void crf_score_kernel(
    const float* __restrict__ emissions,
    const float* __restrict__ transitions,
    const float* __restrict__ start_t,
    const float* __restrict__ end_t,
    const int*   __restrict__ tags,      // [B][S]
    const int*   __restrict__ lengths,   // [B]
    float*       __restrict__ log_num)
{
    const int b = blockIdx.x * blockDim.x + threadIdx.x;
    if (b >= B_) return;
    const int len = lengths[b];
    int tprev = tags[(size_t)b * S_];
    float score = start_t[tprev] + emissions[((size_t)b * S_) * T_ + tprev];
    for (int s = 1; s < S_; ++s) {
        const int tc = tags[(size_t)b * S_ + s];
        if (s < len)
            score += emissions[((size_t)b * S_ + s) * T_ + tc] +
                     transitions[tprev * T_ + tc];
        tprev = tc;
    }
    const int last = tags[(size_t)b * S_ + (len - 1)];
    log_num[b] = score + end_t[last];
}

// ---------------------------------------------------------------------------
// Final: mean(log_den - log_num)
// ---------------------------------------------------------------------------
__global__ void crf_reduce_kernel(const float* __restrict__ log_den,
                                  const float* __restrict__ log_num,
                                  float* __restrict__ out)
{
    __shared__ float sm[B_];
    const int tid = threadIdx.x;
    sm[tid] = log_den[tid] - log_num[tid];
    __syncthreads();
    for (int off = B_ / 2; off > 0; off >>= 1) {
        if (tid < off) sm[tid] += sm[tid + off];
        __syncthreads();
    }
    if (tid == 0) out[0] = sm[0] / (float)B_;
}

extern "C" void kernel_launch(void* const* d_in, const int* in_sizes, int n_in,
                              void* d_out, int out_size, void* d_ws, size_t ws_size,
                              hipStream_t stream)
{
    const float* emissions   = (const float*)d_in[0];
    const float* transitions = (const float*)d_in[1];
    const float* start_t     = (const float*)d_in[2];
    const float* end_t       = (const float*)d_in[3];
    const int*   tags        = (const int*)d_in[4];
    const int*   lengths     = (const int*)d_in[5];

    float* log_den = (float*)d_ws;
    float* log_num = log_den + B_;

    crf_forward_kernel<<<B_ / 16, 256, 0, stream>>>(
        emissions, transitions, start_t, end_t, lengths, log_den);
    crf_score_kernel<<<B_ / 256, 256, 0, stream>>>(
        emissions, transitions, start_t, end_t, tags, lengths, log_num);
    crf_reduce_kernel<<<1, B_, 0, stream>>>(log_den, log_num, (float*)d_out);
}
